// EdgeFeatureGCN_43782896615864
// MI455X (gfx1250) — matrix operand order repaired and checked by reading, not compiled, not run on hardware
//
#include <hip/hip_runtime.h>
#include <hip/hip_bf16.h>
#include <stdint.h>

typedef __attribute__((ext_vector_type(16))) _Float16 v16h;
typedef __attribute__((ext_vector_type(8)))  _Float16 v8h;
typedef __attribute__((ext_vector_type(8)))  float    v8f;

#define N_NODES 10000
#define N_EDGES 80000
#define EP      (N_EDGES + N_NODES)   // 90000 edges incl. self loops
#define NH      4
#define CH      512
#define HC      (NH * CH)             // 2048
#define IN_F    1242
#define KP0     1280                  // IN_F padded to 64 -> even k-tile count
#define KT0     (KP0 / 32)            // 40 k-tiles, layer 0 (even)
#define KT12    (CH / 32)             // 16 k-tiles, layers 1-2 (even)
#define MT      626                   // ceil(10000/16)=625 -> even 626
#define NT      (HC / 16)             // 128 n-tiles
#define OUT_SZ  372600
#define NEG_SLOPE 0.2f
#define EPS_GN    1e-5f

// ---------------- utility fills ----------------
__global__ void k_fill_f32(float* __restrict__ p, int n, float v) {
    int i = blockIdx.x * blockDim.x + threadIdx.x;
    if (i < n) p[i] = v;
}
__global__ void k_fill_i32(int* __restrict__ p, int n, int v) {
    int i = blockIdx.x * blockDim.x + threadIdx.x;
    if (i < n) p[i] = v;
}

// ---------------- monotone float<->int key for atomicMax on floats ----------
__device__ __forceinline__ int enc_f(float f) {
    int b = __float_as_int(f);
    return b ^ ((b >> 31) & 0x7FFFFFFF);
}
__device__ __forceinline__ float dec_f(int k) {
    return __int_as_float(k ^ ((k >> 31) & 0x7FFFFFFF));
}

// ---------------- pack A [M,K] f32 -> WMMA f16 A-fragment tiles -------------
// 16-bit A 16x32: lane<16 holds M=lane,   K in {0..7} U {16..23}
//                 lane>=16 holds M=lane-16,K in {8..15} U {24..31}
// tile = 512 halves; lane L: halves [L*8..L*8+8) (VGPR0-3) and [256+L*8..) (VGPR4-7)
__global__ __launch_bounds__(256)
void k_pack_a(const float* __restrict__ A, _Float16* __restrict__ Ap,
              int Mrows, int K, int KT) {
    int gid = blockIdx.x * 256 + threadIdx.x;
    int tiles = MT * KT;
    int tile = gid >> 6;
    if (tile >= tiles) return;
    int pair = gid & 63;
    int L = pair & 31, g = pair >> 5;
    int mt = tile / KT, kt = tile % KT;
    int m = mt * 16 + (L & 15);
    int kbase = kt * 32 + ((L & 16) ? 8 : 0) + g * 16;
    _Float16* dst = Ap + (size_t)tile * 512 + (size_t)(g * 32 + L) * 8;
    const float* srcp = A + (size_t)m * K + kbase;
#pragma unroll
    for (int j = 0; j < 8; ++j) {
        int k = kbase + j;
        float v = (m < Mrows && k < K) ? srcp[j] : 0.0f;
        dst[j] = (_Float16)v;
    }
}

// ---------------- pack W [K, HC] f32 -> WMMA f16 B-fragment tiles -----------
// 16-bit B 32x16: lane<16 holds N=lane,    K in {0..7} U {16..23}
//                 lane>=16 holds N=lane-16, K in {8..15} U {24..31}
__global__ __launch_bounds__(256)
void k_pack_b(const float* __restrict__ W, _Float16* __restrict__ Bp,
              int K, int KT) {
    int gid = blockIdx.x * 256 + threadIdx.x;
    int tiles = NT * KT;
    int tile = gid >> 6;
    if (tile >= tiles) return;
    int pair = gid & 63;
    int L = pair & 31, g = pair >> 5;
    int nt = tile / KT, kt = tile % KT;
    int n = nt * 16 + (L & 15);
    int kbase = kt * 32 + ((L & 16) ? 8 : 0) + g * 16;
    _Float16* dst = Bp + (size_t)tile * 512 + (size_t)(g * 32 + L) * 8;
#pragma unroll
    for (int j = 0; j < 8; ++j) {
        int k = kbase + j;
        float v = (k < K) ? W[(size_t)k * HC + n] : 0.0f;
        dst[j] = (_Float16)v;
    }
}

__device__ __forceinline__ v16h load_frag(const _Float16* __restrict__ tile, int lane) {
    v8h lo = *(const v8h*)(tile + lane * 8);
    v8h hi = *(const v8h*)(tile + 256 + lane * 8);
    return __builtin_shufflevector(lo, hi, 0, 1, 2, 3, 4, 5, 6, 7,
                                   8, 9, 10, 11, 12, 13, 14, 15);
}

// ---------------- WMMA GEMM: C[M,HC] = Apacked * Bpacked --------------------
// block = 8 waves; block tile = 32 rows x 256 cols; wave tile = 16 x 64.
// Ping-pong register double-buffering, unrolled by 2 k-tiles: fragment sets P
// and Q alternate, each refilled directly from global loads (no register
// copies), so loads for one k-step are in flight under the WMMAs of the other.
// Requires even KT (guaranteed by padding).
__global__ __launch_bounds__(256)
void k_wmma_gemm(const _Float16* __restrict__ Ap, const _Float16* __restrict__ Bp,
                 float* __restrict__ Cmat, int KT) {
    const int lane = threadIdx.x & 31;
    const int w    = threadIdx.x >> 5;
    const int mt   = blockIdx.x * 2 + (w & 1);
    const int ntb  = blockIdx.y * 16 + (w >> 1) * 4;

    const _Float16* ap = Ap + (size_t)mt * KT * 512;
    v8f acc[4] = {};

    // prologue: fragment set P <- k-tile 0
    v16h aP = load_frag(ap, lane);
    v16h bP0 = load_frag(Bp + ((size_t)(ntb + 0) * KT) * 512, lane);
    v16h bP1 = load_frag(Bp + ((size_t)(ntb + 1) * KT) * 512, lane);
    v16h bP2 = load_frag(Bp + ((size_t)(ntb + 2) * KT) * 512, lane);
    v16h bP3 = load_frag(Bp + ((size_t)(ntb + 3) * KT) * 512, lane);

    for (int kt = 0; kt < KT; kt += 2) {
        // fragment set Q <- k-tile kt+1 (valid: KT even)
        const int k1 = kt + 1;
        v16h aQ  = load_frag(ap + (size_t)k1 * 512, lane);
        v16h bQ0 = load_frag(Bp + ((size_t)(ntb + 0) * KT + k1) * 512, lane);
        v16h bQ1 = load_frag(Bp + ((size_t)(ntb + 1) * KT + k1) * 512, lane);
        v16h bQ2 = load_frag(Bp + ((size_t)(ntb + 2) * KT + k1) * 512, lane);
        v16h bQ3 = load_frag(Bp + ((size_t)(ntb + 3) * KT + k1) * 512, lane);

        // WMMAs for k-tile kt (set P) while Q loads are in flight
        acc[0] = __builtin_amdgcn_wmma_f32_16x16x32_f16(false, aP, false, bP0, (short)0, acc[0], false, false);
        acc[1] = __builtin_amdgcn_wmma_f32_16x16x32_f16(false, aP, false, bP1, (short)0, acc[1], false, false);
        acc[2] = __builtin_amdgcn_wmma_f32_16x16x32_f16(false, aP, false, bP2, (short)0, acc[2], false, false);
        acc[3] = __builtin_amdgcn_wmma_f32_16x16x32_f16(false, aP, false, bP3, (short)0, acc[3], false, false);

        // refill set P <- k-tile kt+2 (clamped on final iteration; redundant
        // loads are harmless, avoids a branch and any register copies)
        const int k2 = (kt + 2 < KT) ? (kt + 2) : kt;
        aP  = load_frag(ap + (size_t)k2 * 512, lane);
        bP0 = load_frag(Bp + ((size_t)(ntb + 0) * KT + k2) * 512, lane);
        bP1 = load_frag(Bp + ((size_t)(ntb + 1) * KT + k2) * 512, lane);
        bP2 = load_frag(Bp + ((size_t)(ntb + 2) * KT + k2) * 512, lane);
        bP3 = load_frag(Bp + ((size_t)(ntb + 3) * KT + k2) * 512, lane);

        // WMMAs for k-tile kt+1 (set Q) while P loads are in flight
        acc[0] = __builtin_amdgcn_wmma_f32_16x16x32_f16(false, aQ, false, bQ0, (short)0, acc[0], false, false);
        acc[1] = __builtin_amdgcn_wmma_f32_16x16x32_f16(false, aQ, false, bQ1, (short)0, acc[1], false, false);
        acc[2] = __builtin_amdgcn_wmma_f32_16x16x32_f16(false, aQ, false, bQ2, (short)0, acc[2], false, false);
        acc[3] = __builtin_amdgcn_wmma_f32_16x16x32_f16(false, aQ, false, bQ3, (short)0, acc[3], false, false);
    }

    // C/D layout: VGPR r -> M = r + (lane>=16 ? 8 : 0); N = lane & 15
    const int mbase = mt * 16 + ((lane & 16) ? 8 : 0);
    const int ncol  = lane & 15;
#pragma unroll
    for (int j = 0; j < 4; ++j) {
        int n = (ntb + j) * 16 + ncol;
#pragma unroll
        for (int r = 0; r < 8; ++r) {
            int m = mbase + r;
            if (m < N_NODES) Cmat[(size_t)m * HC + n] = acc[j][r];
        }
    }
}

// ---------------- attention logits: a_s/a_d[n,h] = <H[n,h,:], att[h,:]> ----
__global__ __launch_bounds__(256)
void k_att_dots(const float* __restrict__ Hbuf, const float* __restrict__ att_s,
                const float* __restrict__ att_d, float* __restrict__ a_s,
                float* __restrict__ a_d) {
    int w = threadIdx.x >> 5, lane = threadIdx.x & 31;
    int idx = blockIdx.x * 8 + w;            // (n, h) flat, one wave each
    if (idx >= N_NODES * NH) return;
    int n = idx >> 2, hh = idx & 3;
    const float* hp = Hbuf + (size_t)n * HC + hh * CH;
    const float* sp = att_s + hh * CH;
    const float* dp = att_d + hh * CH;
    float ps = 0.f, pd = 0.f;
#pragma unroll 4
    for (int c = lane; c < CH; c += 32) {
        float hv = hp[c];
        ps += hv * sp[c];
        pd += hv * dp[c];
    }
#pragma unroll
    for (int off = 16; off > 0; off >>= 1) {
        ps += __shfl_xor(ps, off, 32);
        pd += __shfl_xor(pd, off, 32);
    }
    if (lane == 0) { a_s[idx] = ps; a_d[idx] = pd; }
}

// ---------------- edge pass 1: e = leaky_relu(a_s[src]+a_d[dst]); max ------
__global__ void k_edge_max(const int* __restrict__ ei, const float* __restrict__ a_s,
                           const float* __restrict__ a_d, float* __restrict__ e_buf,
                           int* __restrict__ emax) {
    int i = blockIdx.x * blockDim.x + threadIdx.x;
    if (i >= EP) return;
    int s = (i < N_EDGES) ? ei[i] : (i - N_EDGES);
    int d = (i < N_EDGES) ? ei[N_EDGES + i] : (i - N_EDGES);
    float4 as4 = *(const float4*)(a_s + (size_t)s * 4);
    float4 ad4 = *(const float4*)(a_d + (size_t)d * 4);
    float e0 = as4.x + ad4.x; e0 = (e0 >= 0.f) ? e0 : e0 * NEG_SLOPE;
    float e1 = as4.y + ad4.y; e1 = (e1 >= 0.f) ? e1 : e1 * NEG_SLOPE;
    float e2 = as4.z + ad4.z; e2 = (e2 >= 0.f) ? e2 : e2 * NEG_SLOPE;
    float e3 = as4.w + ad4.w; e3 = (e3 >= 0.f) ? e3 : e3 * NEG_SLOPE;
    *(float4*)(e_buf + (size_t)i * 4) = make_float4(e0, e1, e2, e3);
    atomicMax(&emax[d * 4 + 0], enc_f(e0));
    atomicMax(&emax[d * 4 + 1], enc_f(e1));
    atomicMax(&emax[d * 4 + 2], enc_f(e2));
    atomicMax(&emax[d * 4 + 3], enc_f(e3));
}

// ---------------- edge pass 2: ex = exp(e - emax[dst]); denom sum ----------
__global__ void k_edge_exp(const int* __restrict__ ei, const float* __restrict__ e_buf,
                           const int* __restrict__ emax, float* __restrict__ ex_buf,
                           float* __restrict__ denom) {
    int i = blockIdx.x * blockDim.x + threadIdx.x;
    if (i >= EP) return;
    int d = (i < N_EDGES) ? ei[N_EDGES + i] : (i - N_EDGES);
    float4 e4 = *(const float4*)(e_buf + (size_t)i * 4);
    int4 k4 = *(const int4*)(emax + (size_t)d * 4);
    float x0 = __expf(e4.x - dec_f(k4.x));
    float x1 = __expf(e4.y - dec_f(k4.y));
    float x2 = __expf(e4.z - dec_f(k4.z));
    float x3 = __expf(e4.w - dec_f(k4.w));
    *(float4*)(ex_buf + (size_t)i * 4) = make_float4(x0, x1, x2, x3);
    atomicAdd(&denom[d * 4 + 0], x0);
    atomicAdd(&denom[d * 4 + 1], x1);
    atomicAdd(&denom[d * 4 + 2], x2);
    atomicAdd(&denom[d * 4 + 3], x3);
}

// ---------------- edge pass 3: agg[dst] += alpha * H[src] ------------------
// one block per edge; thread t owns 8 contiguous channels (same head)
__global__ __launch_bounds__(256)
void k_edge_agg(const int* __restrict__ ei, const float* __restrict__ ex_buf,
                const float* __restrict__ denom, const float* __restrict__ Hbuf,
                float* __restrict__ agg) {
    int i = blockIdx.x;
    int s = (i < N_EDGES) ? ei[i] : (i - N_EDGES);
    int d = (i < N_EDGES) ? ei[N_EDGES + i] : (i - N_EDGES);
    int hc = threadIdx.x * 8;
    int hh = hc >> 9;
    float alpha = ex_buf[(size_t)i * 4 + hh] / (denom[d * 4 + hh] + 1e-16f);
    const float4* hp = (const float4*)(Hbuf + (size_t)s * HC + hc);
    float4 v0 = hp[0], v1 = hp[1];
    float* ap = agg + (size_t)d * HC + hc;
    atomicAdd(ap + 0, alpha * v0.x);
    atomicAdd(ap + 1, alpha * v0.y);
    atomicAdd(ap + 2, alpha * v0.z);
    atomicAdd(ap + 3, alpha * v0.w);
    atomicAdd(ap + 4, alpha * v1.x);
    atomicAdd(ap + 5, alpha * v1.y);
    atomicAdd(ap + 6, alpha * v1.z);
    atomicAdd(ap + 7, alpha * v1.w);
}

// ---------------- head mean + bias -----------------------------------------
__global__ void k_head_mean(const float* __restrict__ agg, const float* __restrict__ bias,
                            float* __restrict__ hcur) {
    int idx = blockIdx.x * blockDim.x + threadIdx.x;
    if (idx >= N_NODES * CH) return;
    int n = idx >> 9, c = idx & 511;
    const float* ap = agg + (size_t)n * HC + c;
    float s = ap[0] + ap[CH] + ap[2 * CH] + ap[3 * CH];
    hcur[idx] = 0.25f * s + bias[c];
}

// ---------------- column reductions (GraphNorm / pooling) ------------------
__global__ __launch_bounds__(256)
void k_colsum(const float* __restrict__ x, float* __restrict__ colsum, int rows) {
    int c = blockIdx.x * 256 + threadIdx.x;     // 2 blocks in x cover 512 ch
    int chunk = (rows + gridDim.y - 1) / gridDim.y;
    int r0 = blockIdx.y * chunk;
    int r1 = r0 + chunk; if (r1 > rows) r1 = rows;
    float s = 0.f;
#pragma unroll 4
    for (int r = r0; r < r1; ++r) s += x[(size_t)r * CH + c];
    atomicAdd(&colsum[c], s);
}

__global__ __launch_bounds__(256)
void k_colvar(const float* __restrict__ x, const float* __restrict__ colsum,
              const float* __restrict__ ms, float* __restrict__ varsum, int rows) {
    int c = blockIdx.x * 256 + threadIdx.x;
    float shift = (colsum[c] / (float)rows) * ms[c];
    int chunk = (rows + gridDim.y - 1) / gridDim.y;
    int r0 = blockIdx.y * chunk;
    int r1 = r0 + chunk; if (r1 > rows) r1 = rows;
    float s = 0.f;
#pragma unroll 4
    for (int r = r0; r < r1; ++r) {
        float dlt = x[(size_t)r * CH + c] - shift;
        s += dlt * dlt;
    }
    atomicAdd(&varsum[c], s);
}

__global__ void k_gn_apply(float* __restrict__ x, const float* __restrict__ colsum,
                           const float* __restrict__ varsum, const float* __restrict__ ms,
                           const float* __restrict__ w, const float* __restrict__ b) {
    int idx = blockIdx.x * blockDim.x + threadIdx.x;
    if (idx >= N_NODES * CH) return;
    int c = idx & 511;
    float mean = colsum[c] * (1.0f / N_NODES);
    float xc = x[idx] - mean * ms[c];
    float var = varsum[c] * (1.0f / N_NODES);
    float y = xc * rsqrtf(var + EPS_GN) * w[c] + b[c];
    x[idx] = fmaxf(y, 0.f);
}

// ---------------- MLP head --------------------------------------------------
__global__ void k_mlp1(const float* __restrict__ colsum, const float* __restrict__ ow1,
                       const float* __restrict__ ob1, float* __restrict__ z) {
    int c = blockIdx.x * blockDim.x + threadIdx.x;
    if (c >= CH) return;
    float acc = ob1[c];
#pragma unroll 4
    for (int k = 0; k < CH; ++k)
        acc += (colsum[k] * (1.0f / N_NODES)) * ow1[(size_t)k * CH + c];
    z[c] = fmaxf(acc, 0.f);
}

__global__ __launch_bounds__(256)
void k_mlp2(const float* __restrict__ z, const float* __restrict__ ow2,
            const float* __restrict__ ob2, float* __restrict__ out) {
    __shared__ float zs[CH];
    for (int c = threadIdx.x; c < CH; c += 256) zs[c] = z[c];
    __syncthreads();
    int o = blockIdx.x * 256 + threadIdx.x;
    if (o >= OUT_SZ) return;
    float acc = ob2[o];
#pragma unroll 8
    for (int k = 0; k < CH; ++k) acc += zs[k] * ow2[(size_t)k * OUT_SZ + o];
    out[o] = acc;
}

// ---------------- host orchestration ----------------------------------------
extern "C" void kernel_launch(void* const* d_in, const int* in_sizes, int n_in,
                              void* d_out, int out_size, void* d_ws, size_t ws_size,
                              hipStream_t stream) {
    (void)in_sizes; (void)n_in; (void)out_size; (void)ws_size;
    const float* x  = (const float*)d_in[0];
    const int*   ei = (const int*)d_in[1];
    const float* Ws[3]   = {(const float*)d_in[3],  (const float*)d_in[4],  (const float*)d_in[5]};
    const float* atS[3]  = {(const float*)d_in[6],  (const float*)d_in[7],  (const float*)d_in[8]};
    const float* atD[3]  = {(const float*)d_in[9],  (const float*)d_in[10], (const float*)d_in[11]};
    const float* bias[3] = {(const float*)d_in[12], (const float*)d_in[13], (const float*)d_in[14]};
    const float* gnw[3]  = {(const float*)d_in[15], (const float*)d_in[16], (const float*)d_in[17]};
    const float* gnb[3]  = {(const float*)d_in[18], (const float*)d_in[19], (const float*)d_in[20]};
    const float* gnm[3]  = {(const float*)d_in[21], (const float*)d_in[22], (const float*)d_in[23]};
    const float* ow1 = (const float*)d_in[24];
    const float* ob1 = (const float*)d_in[25];
    const float* ow2 = (const float*)d_in[26];
    const float* ob2 = (const float*)d_in[27];

    char* ws = (char*)d_ws;
    size_t off = 0;
    auto alloc = [&](size_t bytes) -> char* {
        size_t o = (off + 255) & ~(size_t)255;
        off = o + bytes;
        return ws + o;
    };
    _Float16* Ap   = (_Float16*)alloc((size_t)MT * KT0 * 512 * sizeof(_Float16));
    _Float16* Bp   = (_Float16*)alloc((size_t)NT * KT0 * 512 * sizeof(_Float16));
    float* Hbuf    = (float*)alloc((size_t)N_NODES * HC * sizeof(float));
    float* agg     = (float*)alloc((size_t)N_NODES * HC * sizeof(float));
    float* hcur    = (float*)alloc((size_t)N_NODES * CH * sizeof(float));
    float* a_s     = (float*)alloc((size_t)N_NODES * NH * sizeof(float));
    float* a_d     = (float*)alloc((size_t)N_NODES * NH * sizeof(float));
    float* e_buf   = (float*)alloc((size_t)EP * NH * sizeof(float));
    float* ex_buf  = (float*)alloc((size_t)EP * NH * sizeof(float));
    int*   emax    = (int*)alloc((size_t)N_NODES * NH * sizeof(int));
    float* denom   = (float*)alloc((size_t)N_NODES * NH * sizeof(float));
    float* colsum  = (float*)alloc(CH * sizeof(float));
    float* varsum  = (float*)alloc(CH * sizeof(float));
    float* z       = (float*)alloc(CH * sizeof(float));

    for (int l = 0; l < 3; ++l) {
        const float* Ain = (l == 0) ? x : hcur;
        int K  = (l == 0) ? IN_F : CH;
        int KT = (l == 0) ? KT0 : KT12;

        int ta = MT * KT * 64;
        k_pack_a<<<(ta + 255) / 256, 256, 0, stream>>>(Ain, Ap, N_NODES, K, KT);
        int tb = NT * KT * 64;
        k_pack_b<<<(tb + 255) / 256, 256, 0, stream>>>(Ws[l], Bp, K, KT);

        dim3 gg(MT / 2, NT / 16);
        k_wmma_gemm<<<gg, 256, 0, stream>>>(Ap, Bp, Hbuf, KT);

        k_att_dots<<<(N_NODES * NH) / 8, 256, 0, stream>>>(Hbuf, atS[l], atD[l], a_s, a_d);

        k_fill_i32<<<(N_NODES * NH + 255) / 256, 256, 0, stream>>>(emax, N_NODES * NH, (int)0x80000000);
        k_fill_f32<<<(N_NODES * NH + 255) / 256, 256, 0, stream>>>(denom, N_NODES * NH, 0.f);
        k_fill_f32<<<(N_NODES * HC + 255) / 256, 256, 0, stream>>>(agg, N_NODES * HC, 0.f);

        k_edge_max<<<(EP + 255) / 256, 256, 0, stream>>>(ei, a_s, a_d, e_buf, emax);
        k_edge_exp<<<(EP + 255) / 256, 256, 0, stream>>>(ei, e_buf, emax, ex_buf, denom);
        k_edge_agg<<<EP, 256, 0, stream>>>(ei, ex_buf, denom, Hbuf, agg);

        k_head_mean<<<(N_NODES * CH + 255) / 256, 256, 0, stream>>>(agg, bias[l], hcur);

        k_fill_f32<<<2, 256, 0, stream>>>(colsum, CH, 0.f);
        k_fill_f32<<<2, 256, 0, stream>>>(varsum, CH, 0.f);
        k_colsum<<<dim3(2, 32), 256, 0, stream>>>(hcur, colsum, N_NODES);
        k_colvar<<<dim3(2, 32), 256, 0, stream>>>(hcur, colsum, gnm[l], varsum, N_NODES);
        k_gn_apply<<<(N_NODES * CH + 255) / 256, 256, 0, stream>>>(hcur, colsum, varsum,
                                                                   gnm[l], gnw[l], gnb[l]);
    }

    // global mean pool + MLP
    k_fill_f32<<<2, 256, 0, stream>>>(colsum, CH, 0.f);
    k_colsum<<<dim3(2, 32), 256, 0, stream>>>(hcur, colsum, N_NODES);
    k_mlp1<<<2, 256, 0, stream>>>(colsum, ow1, ob1, z);
    k_mlp2<<<(OUT_SZ + 255) / 256, 256, 0, stream>>>(z, ow2, ob2, (float*)d_out);
}
